// SwinDiTAttention_50972671869514
// MI455X (gfx1250) — compile-verified, matrix operand order checked by reading, not run on hardware
//
#include <hip/hip_runtime.h>
#include <hip/hip_bf16.h>

typedef __bf16 bf16;
typedef bf16 v16bf __attribute__((ext_vector_type(16)));
typedef bf16 v8bf  __attribute__((ext_vector_type(8)));
typedef float v8f  __attribute__((ext_vector_type(8)));

#define DIMC 256
#define NTOK 64
#define NWIN 256
#define WPACK_ELEMS 65536   // 256x256 bf16 per weight, fragment-major

// workspace layout (bytes)
#define RPB_OFF   0          // f32 [8][64][64] = 131072
#define SCALE_OFF 131072     // f32 [8]
#define WPACK_OFF 131104     // bf16, 4 * 131072 bytes

// LDS layout (bytes), dynamic shared
#define SX_OFF   0           // bf16 [64][264]
#define SQ_OFF   33792       // f32  [64][264]
#define SK_OFF   101376      // f32  [64][264]
#define SVT_OFF  168960      // bf16 [256][72]   (V transposed: [d][token])
#define SQB_OFF  205824      // bf16 [64][264]
#define SKB_OFF  239616      // bf16 [64][264]
#define SP_OFF   SQ_OFF      // overlay: per-wave bf16 [64][64] (8192 B each)
#define SC_OFF   SK_OFF      // overlay: bf16 [64][264]
#define SXS_OFF  SQ_OFF      // overlay: f32 [64][256] async staging of X (64 KB)
#define SMEM_BYTES 273408

__device__ __forceinline__ unsigned short f2bf(float f) {
  unsigned int u = __float_as_uint(f);
  u += 0x7FFFu + ((u >> 16) & 1u);   // round-to-nearest-even
  return (unsigned short)(u >> 16);
}

__device__ __forceinline__ v16bf cat8(v8bf lo, v8bf hi) {
  return __builtin_shufflevector(lo, hi, 0,1,2,3,4,5,6,7,8,9,10,11,12,13,14,15);
}

// A fragment (16x32 bf16) from row-major LDS: lane16 = row, kh selects K-half.
__device__ __forceinline__ v16bf load_afrag(const unsigned short* base) {
  return cat8(*(const v8bf*)base, *(const v8bf*)(base + 16));
}

// M=64 x N=32 x K=256 GEMM: A from LDS bf16 (row-major), B from packed global bf16.
__device__ __forceinline__ void gemm64x32(
    const unsigned short* __restrict__ sA, int strideA,
    const unsigned short* __restrict__ wpack, int ntBase,
    int lane16, int kh, float bias0, float bias1,
    v8f acc[4][2])
{
#pragma unroll
  for (int mt = 0; mt < 4; ++mt)
#pragma unroll
    for (int e = 0; e < 8; ++e) { acc[mt][0][e] = bias0; acc[mt][1][e] = bias1; }
  const int lane = kh * 16 + lane16;
  for (int kt = 0; kt < 8; ++kt) {
    v16bf a[4];
#pragma unroll
    for (int mt = 0; mt < 4; ++mt)
      a[mt] = load_afrag(sA + (mt*16 + lane16)*strideA + kt*32 + kh*8);
    v16bf bfr[2];
#pragma unroll
    for (int j = 0; j < 2; ++j)
      bfr[j] = *(const v16bf*)(wpack + (((kt*16 + ntBase + j)*32 + lane) << 4));
    if (kt < 7)   // pull next weight panel toward the WGP caches
      __builtin_prefetch(wpack + ((((kt+1)*16 + ntBase)*32 + lane) << 4), 0, 3);
#pragma unroll
    for (int mt = 0; mt < 4; ++mt)
#pragma unroll
      for (int j = 0; j < 2; ++j)
        acc[mt][j] = __builtin_amdgcn_wmma_f32_16x16x32_bf16(
            false, a[mt], false, bfr[j], (short)0, acc[mt][j], false, false);
  }
}

// ---------------- kernel 0: pack weights into bf16 B-fragment-major layout ----
extern "C" __global__ __launch_bounds__(256)
void pack_weights(const float* __restrict__ Wq, const float* __restrict__ Wk,
                  const float* __restrict__ Wv, const float* __restrict__ Wo,
                  unsigned short* __restrict__ wpack)
{
  const int total = 4 * WPACK_ELEMS;
  for (int i = blockIdx.x*blockDim.x + threadIdx.x; i < total;
       i += gridDim.x*blockDim.x) {
    int widx = i >> 16;
    int t    = i & 65535;
    int e    = t & 15;
    int lane = (t >> 4) & 31;
    int nt   = (t >> 9) & 15;
    int kt   = (t >> 13) & 7;
    int n    = nt*16 + (lane & 15);
    int khh  = lane >> 4;
    int ke   = (e < 8) ? (khh*8 + e) : (16 + khh*8 + (e - 8));
    int k    = kt*32 + ke;
    const float* src = (widx == 0) ? Wq : (widx == 1) ? Wk : (widx == 2) ? Wv : Wo;
    wpack[i] = f2bf(src[k*DIMC + n]);
  }
}

// ---------------- kernel 1: CPB MLP -> relative position bias + head scales ---
extern "C" __global__ __launch_bounds__(256)
void build_rpb(const float* __restrict__ w1, const float* __restrict__ b1,
               const float* __restrict__ w2, const float* __restrict__ lsc,
               float* __restrict__ rpb, float* __restrict__ scales)
{
  __shared__ float bias[225 * 8];
  const int tid = threadIdx.x;
  if (tid < 225) {
    int a = tid / 15, c = tid % 15;
    float ra = (float)(a - 7) * (8.0f / 7.0f);
    float rb = (float)(c - 7) * (8.0f / 7.0f);
    float t0 = copysignf(log2f(fabsf(ra) + 1.0f) * (1.0f/3.0f), ra);
    float t1 = copysignf(log2f(fabsf(rb) + 1.0f) * (1.0f/3.0f), rb);
    float acc[8];
#pragma unroll
    for (int h = 0; h < 8; ++h) acc[h] = 0.f;
    for (int cc = 0; cc < 512; ++cc) {
      float hh = fmaxf(t0 * w1[cc] + t1 * w1[512 + cc] + b1[cc], 0.f);
#pragma unroll
      for (int h = 0; h < 8; ++h) acc[h] += hh * w2[cc*8 + h];
    }
#pragma unroll
    for (int h = 0; h < 8; ++h) bias[tid*8 + h] = acc[h];
  }
  if (tid < 8) scales[tid] = __expf(fminf(lsc[tid], 4.6051702f));  // ln(100)
  __syncthreads();
  for (int i = tid; i < 8*64*64; i += 256) {
    int h   = i >> 12;
    int rem = i & 4095;
    int t1i = rem >> 6, t2 = rem & 63;
    int i1 = t1i >> 3, j1 = t1i & 7, i2 = t2 >> 3, j2 = t2 & 7;
    int rpi = (i1 - i2 + 7)*15 + (j1 - j2 + 7);
    float x = bias[rpi*8 + h];
    rpb[i] = 16.0f / (1.0f + __expf(-x));
  }
}

// ---------------- main fused attention kernel: 1 block = 1 window ------------
extern "C" __global__ __launch_bounds__(256)
void swin_attn_fused(const float* __restrict__ hs,
                     const float* __restrict__ amask,
                     const float* __restrict__ bq,
                     const float* __restrict__ bv,
                     const float* __restrict__ bo,
                     const unsigned short* __restrict__ wpack,
                     const float* __restrict__ rpb,
                     const float* __restrict__ scales,
                     float* __restrict__ out)
{
  extern __shared__ char smem[];
  unsigned short* sX  = (unsigned short*)(smem + SX_OFF);
  float*          sQ  = (float*)(smem + SQ_OFF);
  float*          sK  = (float*)(smem + SK_OFF);
  unsigned short* sVt = (unsigned short*)(smem + SVT_OFF);
  unsigned short* sQb = (unsigned short*)(smem + SQB_OFF);
  unsigned short* sKb = (unsigned short*)(smem + SKB_OFF);
  unsigned short* sC  = (unsigned short*)(smem + SC_OFF);
  float*          sXs = (float*)(smem + SXS_OFF);   // f32 staging overlay

  const int tid    = threadIdx.x;
  const int w      = tid >> 5;          // wave id == head id
  const int lane   = tid & 31;
  const int lane16 = lane & 15;
  const int kh     = lane >> 4;
  const int b      = blockIdx.x;
  const int ntBase = 2 * w;
  const int cb     = 32 * w;            // this head's column base

  // ---- Phase 1: async-stage X[b] (64x256 f32, 64 KB) into LDS, then ---------
  //      convert to bf16. ASYNCcnt-tracked global->LDS DMA, no VGPR round-trip.
  {
    unsigned int ldsoff = (unsigned int)SXS_OFF + (unsigned int)tid * 16u;
    unsigned long long ga =
        (unsigned long long)(const void*)(hs + (size_t)b * (NTOK * DIMC)) +
        (unsigned long long)tid * 16ull;
#pragma unroll
    for (int it = 0; it < 16; ++it) {     // 256 threads x 16B x 16 = 64 KB
      asm volatile("global_load_async_to_lds_b128 %0, %1, off"
                   :: "v"(ldsoff), "v"(ga) : "memory");
      ldsoff += 4096u;
      ga     += 4096ull;
    }
    asm volatile("s_wait_asynccnt 0x0" ::: "memory");
    __syncthreads();
    for (int i = tid; i < NTOK * DIMC / 4; i += 256) {
      float4 f = *(const float4*)(sXs + i * 4);
      int row = i >> 6, c4 = i & 63;
      uint2 pk;
      pk.x = (unsigned int)f2bf(f.x) | ((unsigned int)f2bf(f.y) << 16);
      pk.y = (unsigned int)f2bf(f.z) | ((unsigned int)f2bf(f.w) << 16);
      *(uint2*)(sX + row*264 + c4*4) = pk;
    }
  }
  __syncthreads();

  // ---- Phase 2: Q, K (f32 LDS), V (bf16 transposed LDS) ---------------------
  {
    v8f acc[4][2];
    float b0 = bq[cb + lane16], b1 = bq[cb + 16 + lane16];
    gemm64x32(sX, 264, wpack + 0*WPACK_ELEMS, ntBase, lane16, kh, b0, b1, acc);
#pragma unroll
    for (int mt = 0; mt < 4; ++mt)
#pragma unroll
      for (int j = 0; j < 2; ++j)
#pragma unroll
        for (int r = 0; r < 8; ++r)
          sQ[(mt*16 + r + 8*kh)*264 + cb + j*16 + lane16] = acc[mt][j][r];
  }
  {
    v8f acc[4][2];
    gemm64x32(sX, 264, wpack + 1*WPACK_ELEMS, ntBase, lane16, kh, 0.f, 0.f, acc);
#pragma unroll
    for (int mt = 0; mt < 4; ++mt)
#pragma unroll
      for (int j = 0; j < 2; ++j)
#pragma unroll
        for (int r = 0; r < 8; ++r)
          sK[(mt*16 + r + 8*kh)*264 + cb + j*16 + lane16] = acc[mt][j][r];
  }
  {
    v8f acc[4][2];
    float b0 = bv[cb + lane16], b1 = bv[cb + 16 + lane16];
    gemm64x32(sX, 264, wpack + 2*WPACK_ELEMS, ntBase, lane16, kh, b0, b1, acc);
#pragma unroll
    for (int mt = 0; mt < 4; ++mt)
#pragma unroll
      for (int j = 0; j < 2; ++j)
#pragma unroll
        for (int r = 0; r < 8; ++r)
          sVt[(cb + j*16 + lane16)*72 + mt*16 + r + 8*kh] = f2bf(acc[mt][j][r]);
  }

  // ---- Phase 3a: cosine-normalize q,k for this head (fold logit scale in q) -
  const float sc_h = scales[w];
  for (int rr = lane; rr < 64; rr += 32) {
    const float* qr = sQ + rr*264 + cb;
    const float* kr = sK + rr*264 + cb;
    float sq = 0.f, sk = 0.f;
#pragma unroll
    for (int d = 0; d < 32; ++d) { float a = qr[d], c = kr[d]; sq += a*a; sk += c*c; }
    float iq = sc_h / fmaxf(sqrtf(sq), 1e-12f);
    float ik = 1.0f / fmaxf(sqrtf(sk), 1e-12f);
    unsigned short* qb = sQb + rr*264 + cb;
    unsigned short* kb = sKb + rr*264 + cb;
#pragma unroll
    for (int d = 0; d < 32; ++d) { qb[d] = f2bf(qr[d]*iq); kb[d] = f2bf(kr[d]*ik); }
  }
  __syncthreads();   // sQ/sK f32 dead -> overlays become safe

  // ---- Phase 3b: attn = qn @ kn^T + rpb + mask; softmax; P -> LDS bf16 ------
  unsigned short* sP = (unsigned short*)(smem + SP_OFF) + w * (64 * 64);
  {
    v8f att[4][4];
#pragma unroll
    for (int mt = 0; mt < 4; ++mt)
#pragma unroll
      for (int nt = 0; nt < 4; ++nt)
#pragma unroll
        for (int e = 0; e < 8; ++e) att[mt][nt][e] = 0.f;

    v16bf aq[4], bk[4];
#pragma unroll
    for (int mt = 0; mt < 4; ++mt)
      aq[mt] = load_afrag(sQb + (mt*16 + lane16)*264 + cb + kh*8);
#pragma unroll
    for (int nt = 0; nt < 4; ++nt)
      bk[nt] = load_afrag(sKb + (nt*16 + lane16)*264 + cb + kh*8);
#pragma unroll
    for (int mt = 0; mt < 4; ++mt)
#pragma unroll
      for (int nt = 0; nt < 4; ++nt)
        att[mt][nt] = __builtin_amdgcn_wmma_f32_16x16x32_bf16(
            false, aq[mt], false, bk[nt], (short)0, att[mt][nt], false, false);

    const float* rb = rpb + w * 4096;
    const float* mk = amask + (size_t)(b & (NWIN - 1)) * 4096;
#pragma unroll
    for (int mt = 0; mt < 4; ++mt)
#pragma unroll
      for (int nt = 0; nt < 4; ++nt) {
        int base = (mt*16 + 8*kh)*64 + nt*16 + lane16;
#pragma unroll
        for (int r = 0; r < 8; ++r)
          att[mt][nt][r] += rb[base + r*64] + mk[base + r*64];
      }

    // softmax over the key dim: per row = 16 lanes x 4 tiles
#pragma unroll
    for (int mt = 0; mt < 4; ++mt)
#pragma unroll
      for (int r = 0; r < 8; ++r) {
        float mx = fmaxf(fmaxf(att[mt][0][r], att[mt][1][r]),
                         fmaxf(att[mt][2][r], att[mt][3][r]));
#pragma unroll
        for (int off = 1; off < 16; off <<= 1)
          mx = fmaxf(mx, __shfl_xor(mx, off, 32));
        float e0 = __expf(att[mt][0][r] - mx);
        float e1 = __expf(att[mt][1][r] - mx);
        float e2 = __expf(att[mt][2][r] - mx);
        float e3 = __expf(att[mt][3][r] - mx);
        float s = e0 + e1 + e2 + e3;
#pragma unroll
        for (int off = 1; off < 16; off <<= 1)
          s += __shfl_xor(s, off, 32);
        float inv = 1.0f / s;
        unsigned short* pp = sP + (mt*16 + r + 8*kh)*64 + lane16;
        pp[0]  = f2bf(e0 * inv);
        pp[16] = f2bf(e1 * inv);
        pp[32] = f2bf(e2 * inv);
        pp[48] = f2bf(e3 * inv);
      }
  }

  // ---- Phase 3c: ctx = P @ V -> sC (bf16) -----------------------------------
  {
    v8f cacc[4][2];
#pragma unroll
    for (int mt = 0; mt < 4; ++mt)
#pragma unroll
      for (int e = 0; e < 8; ++e) { cacc[mt][0][e] = 0.f; cacc[mt][1][e] = 0.f; }
    for (int kt = 0; kt < 2; ++kt) {
      v16bf ap[4], bvf[2];
#pragma unroll
      for (int mt = 0; mt < 4; ++mt)
        ap[mt] = load_afrag(sP + (mt*16 + lane16)*64 + kt*32 + kh*8);
#pragma unroll
      for (int j = 0; j < 2; ++j)
        bvf[j] = load_afrag(sVt + (cb + j*16 + lane16)*72 + kt*32 + kh*8);
#pragma unroll
      for (int mt = 0; mt < 4; ++mt)
#pragma unroll
        for (int j = 0; j < 2; ++j)
          cacc[mt][j] = __builtin_amdgcn_wmma_f32_16x16x32_bf16(
              false, ap[mt], false, bvf[j], (short)0, cacc[mt][j], false, false);
    }
#pragma unroll
    for (int mt = 0; mt < 4; ++mt)
#pragma unroll
      for (int j = 0; j < 2; ++j)
#pragma unroll
        for (int r = 0; r < 8; ++r)
          sC[(mt*16 + r + 8*kh)*264 + cb + j*16 + lane16] = f2bf(cacc[mt][j][r]);
  }
  __syncthreads();

  // ---- Phase 4: out = ctx @ Wo + bo ----------------------------------------
  {
    v8f acc[4][2];
    float b0 = bo[cb + lane16], b1 = bo[cb + 16 + lane16];
    gemm64x32(sC, 264, wpack + 3*WPACK_ELEMS, ntBase, lane16, kh, b0, b1, acc);
    float* op = out + (size_t)b * (NTOK * DIMC);
#pragma unroll
    for (int mt = 0; mt < 4; ++mt)
#pragma unroll
      for (int j = 0; j < 2; ++j)
#pragma unroll
        for (int r = 0; r < 8; ++r)
          op[(mt*16 + r + 8*kh)*DIMC + cb + j*16 + lane16] = acc[mt][j][r];
  }
}

extern "C" void kernel_launch(void* const* d_in, const int* in_sizes, int n_in,
                              void* d_out, int out_size, void* d_ws, size_t ws_size,
                              hipStream_t stream) {
  (void)in_sizes; (void)n_in; (void)out_size; (void)ws_size;
  const float* hs    = (const float*)d_in[0];
  const float* amask = (const float*)d_in[1];
  const float* Wq    = (const float*)d_in[2];
  const float* bq    = (const float*)d_in[3];
  const float* Wk    = (const float*)d_in[4];
  const float* Wv    = (const float*)d_in[5];
  const float* bv    = (const float*)d_in[6];
  const float* lsc   = (const float*)d_in[7];
  const float* w1    = (const float*)d_in[8];
  const float* b1    = (const float*)d_in[9];
  const float* w2    = (const float*)d_in[10];
  const float* Wo    = (const float*)d_in[11];
  const float* bo    = (const float*)d_in[12];
  float* out = (float*)d_out;

  char* ws = (char*)d_ws;
  float* rpb            = (float*)(ws + RPB_OFF);
  float* scales         = (float*)(ws + SCALE_OFF);
  unsigned short* wpack = (unsigned short*)(ws + WPACK_OFF);

  pack_weights<<<256, 256, 0, stream>>>(Wq, Wk, Wv, Wo, wpack);
  build_rpb<<<1, 256, 0, stream>>>(w1, b1, w2, lsc, rpb, scales);
  swin_attn_fused<<<4096, 256, SMEM_BYTES, stream>>>(
      hs, amask, bq, bv, bo, wpack, rpb, scales, out);
}
